// CurveNet_23536420782440
// MI455X (gfx1250) — matrix operand order repaired and checked by reading, not compiled, not run on hardware
//
#include <hip/hip_runtime.h>
#include <hip/hip_bf16.h>
#include <math.h>

// ---------------------------------------------------------------------------
// CDNA5 (gfx1250) implementation of the CurveNet/PCT reference.
// All GEMMs use v_wmma_f32_16x16x32_f16 (f16 in, f32 accumulate).
// Activations stored token-major [B][N][C] in f16 so every WMMA fragment load
// is a contiguous 16B global load.  BN folded into per-channel alpha/beta.
// ---------------------------------------------------------------------------

typedef __attribute__((ext_vector_type(16))) _Float16 v16h;
typedef __attribute__((ext_vector_type(8)))  float    v8f;

#define DEV static __device__ __forceinline__

constexpr int BATCH = 8;
constexpr int NPTS  = 2048;
constexpr int DDIM  = 128;

union H8  { _Float16 h[8];  float4 v4;     };
union F8  { v8f v;          float  f[8];  float4 v4[2]; };
union H16 { v16h v;         _Float16 h[16]; float4 f4[2]; };

DEV v8f wmma_f16(v16h a, v16h b, v8f c) {
  return __builtin_amdgcn_wmma_f32_16x16x32_f16(false, a, false, b, (short)0, c,
                                                false, false);
}

// A-style fragment: 16 rows (M) x K=32, row-major source, 16b elements.
// lane: row = lane&15, halves j = K ((lane>>4)*16 + j)
DEV v16h frag_rows(const _Float16* base, int stride) {
  int lane = threadIdx.x & 31;
  const _Float16* p = base + (size_t)(lane & 15) * stride + ((lane >> 4) << 4);
  H16 u;
  u.f4[0] = *(const float4*)(p);
  u.f4[1] = *(const float4*)(p + 8);
  return u.v;
}

// B-style fragment: K=32 x 16 cols where B[k,col] = base[col*stride + k].
// lane: col = lane&15, halves j = K ((lane>>4)*16 + j)
DEV v16h frag_cols(const _Float16* base, int stride) {
  int lane = threadIdx.x & 31;
  const _Float16* p = base + (size_t)(lane & 15) * stride + ((lane >> 4) << 4);
  H16 u;
  u.f4[0] = *(const float4*)(p);
  u.f4[1] = *(const float4*)(p + 8);
  return u.v;
}

// B-style fragment from an LDS tile laid out [k][col] with padded stride.
DEV v16h frag_cols_lds(const _Float16* lds, int stride) {
  int lane = threadIdx.x & 31;
  int col = lane & 15;
  int kb  = (lane >> 4) << 4;
  H16 u;
#pragma unroll
  for (int j = 0; j < 16; ++j) u.h[j] = lds[(kb + j) * stride + col];
  return u.v;
}

DEV float activate(float x, int act) {
  if (act == 1) return x > 0.0f ? x : 0.0f;
  if (act == 2) return x > 0.0f ? x : 0.2f * x;
  return x;
}

// ---------------------------------------------------------------------------
// Generic channel GEMM:  per batch  Y = act(alpha*(W @ X) + beta [+betaB]) (+res)
//   X : [B][N][C] f16 token-major,  W : [O][wStride] f16 row-major
// OUT_TRANS=true  -> Y stored [B][N][*] (o contiguous per lane)
// OUT_TRANS=false -> Y stored [B][O][N] (n contiguous per lane), f16 and/or f32
// 8 waves / block, each wave owns a 32x32 tile (4 accs, 4 wmma / K-step).
// ---------------------------------------------------------------------------
template <bool OUT_TRANS>
__global__ void gemm_kernel(const _Float16* __restrict__ X, int C,
                            const _Float16* __restrict__ Wm, int wStride, int O,
                            const float* __restrict__ alpha,
                            const float* __restrict__ beta,
                            const float* __restrict__ betaB, int act,
                            const _Float16* __restrict__ resid,
                            _Float16* __restrict__ out1, int out1Stride,
                            _Float16* __restrict__ out2, int out2Stride,
                            float* __restrict__ out32) {
  const int N = NPTS;
  int g = blockIdx.x * 8 + (threadIdx.x >> 5);
  int ntiles = N / 32, otiles = O / 32;
  if (g >= BATCH * otiles * ntiles) return;
  int nt = g % ntiles;  g /= ntiles;
  int ot = g % otiles;  int b = g / otiles;
  int n0 = nt * 32, o0 = ot * 32;
  const _Float16* Xb = X + (size_t)b * N * C;

  v8f a00 = {0,0,0,0,0,0,0,0}, a01 = a00, a10 = a00, a11 = a00;
  for (int c = 0; c < C; c += 32) {
    v16h r0, r1, c0, c1;
    if (OUT_TRANS) {               // A = W rows (M=o), B cols = X rows (N=n)
      r0 = frag_rows(Wm + (size_t)o0 * wStride + c, wStride);
      r1 = frag_rows(Wm + (size_t)(o0 + 16) * wStride + c, wStride);
      c0 = frag_cols(Xb + (size_t)n0 * C + c, C);
      c1 = frag_cols(Xb + (size_t)(n0 + 16) * C + c, C);
    } else {                       // A = X rows (M=n), B cols = W rows (N=o)
      r0 = frag_rows(Xb + (size_t)n0 * C + c, C);
      r1 = frag_rows(Xb + (size_t)(n0 + 16) * C + c, C);
      c0 = frag_cols(Wm + (size_t)o0 * wStride + c, wStride);
      c1 = frag_cols(Wm + (size_t)(o0 + 16) * wStride + c, wStride);
    }
    a00 = wmma_f16(r0, c0, a00);
    a01 = wmma_f16(r0, c1, a01);
    a10 = wmma_f16(r1, c0, a10);
    a11 = wmma_f16(r1, c1, a11);
  }

  int lane = threadIdx.x & 31;
  v8f accs[2][2] = {{a00, a01}, {a10, a11}};
#pragma unroll
  for (int ig = 0; ig < 2; ++ig)
#pragma unroll
    for (int jg = 0; jg < 2; ++jg) {
      F8 u; u.v = accs[ig][jg];
      if (OUT_TRANS) {
        int n  = n0 + jg * 16 + (lane & 15);
        int ob = o0 + ig * 16 + ((lane >> 4) << 3);
        H8 res;
        if (resid) res.v4 = *(const float4*)(resid + ((size_t)b * N + n) * O + ob);
        H8 oh;
#pragma unroll
        for (int r = 0; r < 8; ++r) {
          int o = ob + r;
          float al = alpha ? alpha[o] : 1.0f;
          float be = (beta ? beta[o] : 0.0f) + (betaB ? betaB[b * O + o] : 0.0f);
          float y = activate(al * u.f[r] + be, act);
          if (resid) y += (float)res.h[r];
          oh.h[r] = (_Float16)y;
        }
        if (out1) *(float4*)(out1 + ((size_t)b * N + n) * out1Stride + ob) = oh.v4;
        if (out2) *(float4*)(out2 + ((size_t)b * N + n) * out2Stride + ob) = oh.v4;
      } else {
        int o  = o0 + jg * 16 + (lane & 15);
        int nb = n0 + ig * 16 + ((lane >> 4) << 3);
        float al = alpha ? alpha[o] : 1.0f;
        float be = (beta ? beta[o] : 0.0f) + (betaB ? betaB[b * O + o] : 0.0f);
        if (out32) {
          F8 of;
#pragma unroll
          for (int r = 0; r < 8; ++r) of.f[r] = activate(al * u.f[r] + be, act);
          float* dst = out32 + ((size_t)b * O + o) * N + nb;
          *(float4*)(dst) = of.v4[0];
          *(float4*)(dst + 4) = of.v4[1];
        }
        if (out1) {
          H8 oh;
#pragma unroll
          for (int r = 0; r < 8; ++r) oh.h[r] = (_Float16)activate(al * u.f[r] + be, act);
          *(float4*)(out1 + ((size_t)b * O + o) * N + nb) = oh.v4;
        }
      }
    }
}

// ---------------------------------------------------------------------------
// Energy: E = Q^T Q per batch (Q stored token-major [N][128]).  Symmetric, so
// D tiles are stored transposed -> contiguous f32 stores.
// ---------------------------------------------------------------------------
__global__ void energy_kernel(const _Float16* __restrict__ Q, float* __restrict__ E) {
  int g = blockIdx.x * 8 + (threadIdx.x >> 5);
  int tiles = NPTS / 32;
  if (g >= tiles * tiles) return;
  int jt = g % tiles, it = g / tiles;
  int i0 = it * 32, j0 = jt * 32;

  v8f a00 = {0,0,0,0,0,0,0,0}, a01 = a00, a10 = a00, a11 = a00;
  for (int c = 0; c < DDIM; c += 32) {
    v16h r0 = frag_rows(Q + (size_t)i0 * DDIM + c, DDIM);
    v16h r1 = frag_rows(Q + (size_t)(i0 + 16) * DDIM + c, DDIM);
    v16h c0 = frag_cols(Q + (size_t)j0 * DDIM + c, DDIM);
    v16h c1 = frag_cols(Q + (size_t)(j0 + 16) * DDIM + c, DDIM);
    a00 = wmma_f16(r0, c0, a00);
    a01 = wmma_f16(r0, c1, a01);
    a10 = wmma_f16(r1, c0, a10);
    a11 = wmma_f16(r1, c1, a11);
  }
  int lane = threadIdx.x & 31;
  v8f accs[2][2] = {{a00, a01}, {a10, a11}};
#pragma unroll
  for (int ig = 0; ig < 2; ++ig)
#pragma unroll
    for (int jg = 0; jg < 2; ++jg) {
      F8 u; u.v = accs[ig][jg];
      int j  = j0 + jg * 16 + (lane & 15);
      int ib = i0 + ig * 16 + ((lane >> 4) << 3);
      float* dst = E + (size_t)j * NPTS + ib;   // E[j][i] == E[i][j] (symmetric)
      *(float4*)(dst) = u.v4[0];
      *(float4*)(dst + 4) = u.v4[1];
    }
}

// Row softmax in place (one block per row).
__global__ void softmax_rows(float* __restrict__ E) {
  __shared__ float red[256];
  float* r = E + (size_t)blockIdx.x * NPTS;
  int t = threadIdx.x;
  float m = -INFINITY;
  for (int j = t; j < NPTS; j += 256) m = fmaxf(m, r[j]);
  red[t] = m; __syncthreads();
  for (int s = 128; s > 0; s >>= 1) { if (t < s) red[t] = fmaxf(red[t], red[t + s]); __syncthreads(); }
  m = red[0]; __syncthreads();
  float vals[NPTS / 256];
  float sum = 0.0f;
#pragma unroll
  for (int i = 0; i < NPTS / 256; ++i) {
    float e = __expf(r[t + i * 256] - m);
    vals[i] = e; sum += e;
  }
  red[t] = sum; __syncthreads();
  for (int s = 128; s > 0; s >>= 1) { if (t < s) red[t] += red[t + s]; __syncthreads(); }
  float inv = 1.0f / red[0];
#pragma unroll
  for (int i = 0; i < NPTS / 256; ++i) r[t + i * 256] = vals[i] * inv;
}

// inv[j] = 1 / (1e-9 + sum_i attn[i][j])
__global__ void colsum_inv(const float* __restrict__ E, float* __restrict__ inv) {
  int j = blockIdx.x * 256 + threadIdx.x;
  float s = 0.0f;
  for (int i = 0; i < NPTS; ++i) s += E[(size_t)i * NPTS + j];
  inv[j] = 1.0f / (1e-9f + s);
}

// ---------------------------------------------------------------------------
// AV: diff[n][d] = x[n][d] - (sum_m V[d][m] * attn[m][n]) * inv[n]
// V is [128][2048] f16; attn tile staged f32->f16 through LDS (the only
// transpose-access in the pipeline).  Block = 8 waves covering 128(d) x 64(n).
// ---------------------------------------------------------------------------
__global__ void av_diff_kernel(const float* __restrict__ E,
                               const float* __restrict__ inv,
                               const _Float16* __restrict__ V,
                               const _Float16* __restrict__ X,
                               _Float16* __restrict__ diff) {
  __shared__ _Float16 tile[32 * 72];
  int wave  = threadIdx.x >> 5;
  int dwave = (wave & 3) * 32;
  int nwave = (wave >> 2) * 32;
  int n0 = blockIdx.x * 64;

  v8f acc[2][2];
  v8f z = {0,0,0,0,0,0,0,0};
  acc[0][0] = z; acc[0][1] = z; acc[1][0] = z; acc[1][1] = z;

  for (int m = 0; m < NPTS; m += 32) {
    __syncthreads();
    for (int t = threadIdx.x; t < 32 * 64; t += 256) {
      int k = t >> 6, col = t & 63;
      tile[k * 72 + col] = (_Float16)E[(size_t)(m + k) * NPTS + n0 + col];
    }
    __syncthreads();
    v16h a0 = frag_rows(V + (size_t)dwave * NPTS + m, NPTS);
    v16h a1 = frag_rows(V + (size_t)(dwave + 16) * NPTS + m, NPTS);
    v16h b0 = frag_cols_lds(tile + nwave, 72);
    v16h b1 = frag_cols_lds(tile + nwave + 16, 72);
    acc[0][0] = wmma_f16(a0, b0, acc[0][0]);
    acc[0][1] = wmma_f16(a0, b1, acc[0][1]);
    acc[1][0] = wmma_f16(a1, b0, acc[1][0]);
    acc[1][1] = wmma_f16(a1, b1, acc[1][1]);
  }

  int lane = threadIdx.x & 31;
#pragma unroll
  for (int ig = 0; ig < 2; ++ig)
#pragma unroll
    for (int jg = 0; jg < 2; ++jg) {
      F8 u; u.v = acc[ig][jg];
      int n  = n0 + nwave + jg * 16 + (lane & 15);
      int db = dwave + ig * 16 + ((lane >> 4) << 3);
      float iv = inv[n];
      H8 xh; xh.v4 = *(const float4*)(X + (size_t)n * DDIM + db);
      H8 oh;
#pragma unroll
      for (int r = 0; r < 8; ++r)
        oh.h[r] = (_Float16)((float)xh.h[r] - u.f[r] * iv);
      *(float4*)(diff + (size_t)n * DDIM + db) = oh.v4;
    }
}

// ---------------------------------------------------------------------------
// Small helper kernels
// ---------------------------------------------------------------------------
__global__ void cvt_f16(const float* __restrict__ s, _Float16* __restrict__ d, int n) {
  int i = blockIdx.x * 256 + threadIdx.x;
  if (i < n) d[i] = (_Float16)s[i];
}

// alpha = g*rsqrt(v+eps); beta = alpha*(bias - m) + b.  g==null -> alpha=1, beta=bias.
__global__ void bn_prep(const float* g, const float* bb, const float* m,
                        const float* v, const float* bias,
                        float* alpha, float* beta, int n) {
  int i = blockIdx.x * 256 + threadIdx.x;
  if (i >= n) return;
  float base = bias ? bias[i] : 0.0f;
  if (g) {
    float a = g[i] * rsqrtf(v[i] + 1e-5f);
    alpha[i] = a;
    beta[i]  = a * (base - m[i]) + bb[i];
  } else {
    alpha[i] = 1.0f;
    beta[i]  = base;
  }
}

// x [B][256][2048] f32  ->  xT [B][2048][256] f16 (32x32 LDS tile transpose)
__global__ void transpose_in(const float* __restrict__ x, _Float16* __restrict__ xt) {
  __shared__ float tile[32][33];
  int bid = blockIdx.x;
  int cb = bid % (256 / 32);  bid /= (256 / 32);
  int nb = bid % (NPTS / 32); int b = bid / (NPTS / 32);
  int tx = threadIdx.x & 31, ty = threadIdx.x >> 5;   // 32 x 8
  const float* xb = x + ((size_t)b * 256 + cb * 32) * NPTS + nb * 32;
#pragma unroll
  for (int i = 0; i < 32; i += 8)
    tile[ty + i][tx] = xb[(size_t)(ty + i) * NPTS + tx];   // tile[c][n]
  __syncthreads();
  _Float16* xtb = xt + ((size_t)b * NPTS + nb * 32) * 256 + cb * 32;
#pragma unroll
  for (int i = 0; i < 32; i += 8)
    xtb[(size_t)(ty + i) * 256 + tx] = (_Float16)tile[tx][ty + i];
}

// max / mean over N for each (b, c) of xf [B][N][1024] -> g [B][2048]
__global__ void pool_kernel(const _Float16* __restrict__ xf, float* __restrict__ g) {
  int c = (blockIdx.x & 3) * 256 + threadIdx.x;
  int b = blockIdx.x >> 2;
  const _Float16* p = xf + (size_t)b * NPTS * 1024 + c;
  float mx = -INFINITY, sm = 0.0f;
  for (int n = 0; n < NPTS; ++n) {
    float v = (float)p[(size_t)n * 1024];
    mx = fmaxf(mx, v); sm += v;
  }
  g[b * 2048 + c] = mx;
  g[b * 2048 + 1024 + c] = sm * (1.0f / NPTS);
}

// betaB[b][o] = alpha_s1[o]*(bs1[o] + Ws1[o,1024:3072]·g[b] - m[o]) + b[o]
__global__ void gvec_kernel(const float* __restrict__ Ws1, const float* __restrict__ bs1,
                            const float* __restrict__ alpha_s1,
                            const float* __restrict__ bn_m, const float* __restrict__ bn_b,
                            const float* __restrict__ g, float* __restrict__ betaB) {
  int o = (blockIdx.x & 1) * 256 + threadIdx.x;   // 512
  int b = blockIdx.x >> 1;
  float s = bs1[o];
  const float* w  = Ws1 + (size_t)o * 3072 + 1024;
  const float* gb = g + b * 2048;
  for (int c = 0; c < 2048; ++c) s += w[c] * gb[c];
  betaB[b * 512 + o] = alpha_s1[o] * (s - bn_m[o]) + bn_b[o];
}

// ---------------------------------------------------------------------------
// Host orchestration
// ---------------------------------------------------------------------------
typedef _Float16 h16;

extern "C" void kernel_launch(void* const* d_in, const int* in_sizes, int n_in,
                              void* d_out, int out_size, void* d_ws, size_t ws_size,
                              hipStream_t stream) {
  (void)in_sizes; (void)n_in; (void)out_size; (void)ws_size;
  // Input order: setup_inputs() insertion order, params flattened depth-first.
  const float* in_x    = (const float*)d_in[0];
  const float* W1      = (const float*)d_in[1];
  const float* bn1[4]  = {(const float*)d_in[2], (const float*)d_in[3], (const float*)d_in[4], (const float*)d_in[5]};
  const float* W2      = (const float*)d_in[6];
  const float* bn2[4]  = {(const float*)d_in[7], (const float*)d_in[8], (const float*)d_in[9], (const float*)d_in[10]};
  const float *Wqk[4], *Wv[4], *bv[4], *Wt[4], *bt[4], *sabn[4][4];
  for (int i = 0; i < 4; ++i) {
    int base = 11 + i * 9;
    Wqk[i] = (const float*)d_in[base + 0];
    Wv[i]  = (const float*)d_in[base + 1];
    bv[i]  = (const float*)d_in[base + 2];
    Wt[i]  = (const float*)d_in[base + 3];
    bt[i]  = (const float*)d_in[base + 4];
    for (int k = 0; k < 4; ++k) sabn[i][k] = (const float*)d_in[base + 5 + k];
  }
  const float* Wfuse   = (const float*)d_in[47];
  const float* bnf[4]  = {(const float*)d_in[48], (const float*)d_in[49], (const float*)d_in[50], (const float*)d_in[51]};
  const float* Ws1     = (const float*)d_in[52];
  const float* bs1     = (const float*)d_in[53];
  const float* bns1[4] = {(const float*)d_in[54], (const float*)d_in[55], (const float*)d_in[56], (const float*)d_in[57]};
  const float* Ws2     = (const float*)d_in[58];
  const float* bs2     = (const float*)d_in[59];
  const float* bns2[4] = {(const float*)d_in[60], (const float*)d_in[61], (const float*)d_in[62], (const float*)d_in[63]};

  // ---- workspace carve ----
  char* wp = (char*)d_ws;
  auto alloc = [&](size_t bytes) -> void* {
    void* r = (void*)wp; wp += (bytes + 255) & ~(size_t)255; return r;
  };
  h16* W1f    = (h16*)alloc(128 * 256 * 2);
  h16* W2f    = (h16*)alloc(128 * 128 * 2);
  h16 *Wqkf[4], *Wvf[4], *Wtf[4];
  for (int i = 0; i < 4; ++i) {
    Wqkf[i] = (h16*)alloc(128 * 128 * 2);
    Wvf[i]  = (h16*)alloc(128 * 128 * 2);
    Wtf[i]  = (h16*)alloc(128 * 128 * 2);
  }
  h16* Wfusef = (h16*)alloc((size_t)1024 * 512 * 2);
  h16* Ws1f   = (h16*)alloc((size_t)512 * 3072 * 2);
  h16* Ws2f   = (h16*)alloc((size_t)256 * 512 * 2);

  float* a1 = (float*)alloc(128 * 4);  float* b1 = (float*)alloc(128 * 4);
  float* a2 = (float*)alloc(128 * 4);  float* b2 = (float*)alloc(128 * 4);
  float *aT[4], *bT[4], *aV[4], *bV[4];
  for (int i = 0; i < 4; ++i) {
    aT[i] = (float*)alloc(128 * 4); bT[i] = (float*)alloc(128 * 4);
    aV[i] = (float*)alloc(128 * 4); bV[i] = (float*)alloc(128 * 4);
  }
  float* af  = (float*)alloc(1024 * 4); float* bf  = (float*)alloc(1024 * 4);
  float* as1 = (float*)alloc(512 * 4);  float* bs1t = (float*)alloc(512 * 4);
  float* as2 = (float*)alloc(256 * 4);  float* bs2v = (float*)alloc(256 * 4);
  float* betaB_s1 = (float*)alloc((size_t)BATCH * 512 * 4);
  float* gpool    = (float*)alloc((size_t)BATCH * 2048 * 4);
  float* inv      = (float*)alloc(NPTS * 4);

  size_t tokF = (size_t)BATCH * NPTS;      // tokens
  h16* xT   = (h16*)alloc(tokF * 256 * 2);
  h16* xA   = (h16*)alloc(tokF * 128 * 2);
  h16* xB   = (h16*)alloc(tokF * 128 * 2);
  h16* qT   = (h16*)alloc(tokF * 128 * 2);
  h16* Vb   = (h16*)alloc((size_t)BATCH * 128 * NPTS * 2);
  h16* diff = (h16*)alloc(tokF * 128 * 2);
  h16* xcat = (h16*)alloc(tokF * 512 * 2);
  h16* xf   = (h16*)alloc(tokF * 1024 * 2);
  h16* xs1  = (h16*)alloc(tokF * 512 * 2);
  float* E  = (float*)alloc((size_t)NPTS * NPTS * 4);

  // ---- weight conversion + BN folding ----
  auto cvt = [&](const float* s, h16* d, int n) {
    cvt_f16<<<(n + 255) / 256, 256, 0, stream>>>(s, d, n);
  };
  cvt(W1, W1f, 128 * 256);
  cvt(W2, W2f, 128 * 128);
  for (int i = 0; i < 4; ++i) {
    cvt(Wqk[i], Wqkf[i], 128 * 128);
    cvt(Wv[i],  Wvf[i],  128 * 128);
    cvt(Wt[i],  Wtf[i],  128 * 128);
  }
  cvt(Wfuse, Wfusef, 1024 * 512);
  cvt(Ws1, Ws1f, 512 * 3072);
  cvt(Ws2, Ws2f, 256 * 512);

  auto prep = [&](const float* const* bn, const float* bias, float* a, float* b, int n) {
    bn_prep<<<(n + 255) / 256, 256, 0, stream>>>(
        bn ? bn[0] : nullptr, bn ? bn[1] : nullptr, bn ? bn[2] : nullptr,
        bn ? bn[3] : nullptr, bias, a, b, n);
  };
  prep(bn1, nullptr, a1, b1, 128);
  prep(bn2, nullptr, a2, b2, 128);
  for (int i = 0; i < 4; ++i) {
    prep(sabn[i], bt[i], aT[i], bT[i], 128);
    prep(nullptr, bv[i], aV[i], bV[i], 128);   // alpha=1, beta=bv
  }
  prep(bnf, nullptr, af, bf, 1024);
  prep(bns1, nullptr, as1, bs1t, 512);
  prep(bns2, bs2, as2, bs2v, 256);

  // ---- input transpose ----
  transpose_in<<<BATCH * (NPTS / 32) * (256 / 32), 256, 0, stream>>>(in_x, xT);

  auto gemmT = [&](const h16* X, int C, const h16* Wm, int wStride, int O,
                   const float* al, const float* be, const float* beB, int act,
                   const h16* res, h16* o1, int o1s, h16* o2, int o2s) {
    int blocks = (BATCH * (O / 32) * (NPTS / 32) + 7) / 8;
    gemm_kernel<true><<<blocks, 256, 0, stream>>>(X, C, Wm, wStride, O, al, be, beB,
                                                  act, res, o1, o1s, o2, o2s, nullptr);
  };
  auto gemmN = [&](const h16* X, int C, const h16* Wm, int wStride, int O,
                   const float* al, const float* be, int act, h16* o1, float* o32) {
    int blocks = (BATCH * (O / 32) * (NPTS / 32) + 7) / 8;
    gemm_kernel<false><<<blocks, 256, 0, stream>>>(X, C, Wm, wStride, O, al, be,
                                                   nullptr, act, nullptr, o1, 0,
                                                   nullptr, 0, o32);
  };

  // ---- stem ----
  gemmT(xT, 256, W1f, 256, 128, a1, b1, nullptr, 1, nullptr, xA, 128, nullptr, 0);
  gemmT(xA, 128, W2f, 128, 128, a2, b2, nullptr, 1, nullptr, xB, 128, nullptr, 0);

  // ---- 4 offset-attention blocks ----
  h16* xcur = xB;
  h16* xnxt = xA;
  for (int i = 0; i < 4; ++i) {
    gemmT(xcur, 128, Wqkf[i], 128, 128, nullptr, nullptr, nullptr, 0, nullptr,
          qT, 128, nullptr, 0);
    gemmN(xcur, 128, Wvf[i], 128, 128, aV[i], bV[i], 0, Vb, nullptr);
    for (int b = 0; b < BATCH; ++b) {
      const h16* Qb = qT + (size_t)b * NPTS * 128;
      energy_kernel<<<((NPTS / 32) * (NPTS / 32) + 7) / 8, 256, 0, stream>>>(Qb, E);
      softmax_rows<<<NPTS, 256, 0, stream>>>(E);
      colsum_inv<<<NPTS / 256, 256, 0, stream>>>(E, inv);
      av_diff_kernel<<<NPTS / 64, 256, 0, stream>>>(
          E, inv, Vb + (size_t)b * 128 * NPTS,
          xcur + (size_t)b * NPTS * 128, diff + (size_t)b * NPTS * 128);
    }
    gemmT(diff, 128, Wtf[i], 128, 128, aT[i], bT[i], nullptr, 1, xcur,
          xnxt, 128, xcat + i * 128, 512);
    h16* t = xcur; xcur = xnxt; xnxt = t;
  }

  // ---- fuse + pooling + head ----
  gemmT(xcat, 512, Wfusef, 512, 1024, af, bf, nullptr, 2, nullptr, xf, 1024, nullptr, 0);
  pool_kernel<<<BATCH * 4, 256, 0, stream>>>(xf, gpool);
  gvec_kernel<<<BATCH * 2, 256, 0, stream>>>(Ws1, bs1, as1, bns1[2], bns1[1],
                                             gpool, betaB_s1);
  gemmT(xf, 1024, Ws1f, 3072, 512, as1, nullptr, betaB_s1, 1, nullptr,
        xs1, 512, nullptr, 0);
  gemmN(xs1, 512, Ws2f, 512, 256, as2, bs2v, 1, nullptr, (float*)d_out);
}
